// GraphAttConvOneHead_44057774522858
// MI455X (gfx1250) — matrix operand order repaired and checked by reading, not compiled, not run on hardware
//
#include <hip/hip_runtime.h>
#include <hip/hip_bf16.h>

typedef __attribute__((ext_vector_type(16))) __bf16 v16bf;
typedef __attribute__((ext_vector_type(8)))  float  v8f;

#define IN_F   256
#define OUT_F  128
#define LRELU  0.2f
#define EPS_F  1e-16f

// -------------------------------------------------------------------------
// init: zero out[] and denom[], set e_max[] to -inf bit pattern
// -------------------------------------------------------------------------
__global__ void gat_init_kernel(float* __restrict__ out, int out_elems,
                                int* __restrict__ emax, float* __restrict__ denom,
                                int n)
{
    int i = blockIdx.x * blockDim.x + threadIdx.x;
    if (i < out_elems) out[i] = 0.0f;
    if (i < n) { emax[i] = (int)0xFF800000u; denom[i] = 0.0f; }
}

// -------------------------------------------------------------------------
// Fused GEMM + attention scores.
//   h = x @ W            via V_WMMA_F32_16X16X32_BF16 (f32 accumulate)
//   sl = h @ a_l, sr = h @ a_r   fused in the epilogue from the live acc[]
// 256 threads = 8 waves; each wave computes a 16(M) x 128(N) strip, block
// covers 128 rows. W staged transposed (col-major, bf16) in LDS in two
// 128-K phases (34KB, padded rows -> conflict-free ds_load_b128).
// All 8 B fragments of a K-chunk are preloaded so the 8 WMMAs issue
// back-to-back behind a single dscnt wait.
// -------------------------------------------------------------------------
__global__ __launch_bounds__(256) void gat_gemm_kernel(
    const float* __restrict__ x, const float* __restrict__ w,
    const float* __restrict__ a,
    float* __restrict__ h, float* __restrict__ sl, float* __restrict__ sr,
    int n)
{
    __shared__ __bf16 ldsW[OUT_F][128 + 8];   // [col][k_local], +8 pad

    const int tid  = threadIdx.x;
    const int wave = tid >> 5;
    const int lane = tid & 31;
    const int m16  = lane & 15;
    const int kh   = lane >> 4;               // K-half select per ISA layout

    const int mBase = blockIdx.x * 128 + wave * 16;
    int rowA = mBase + m16;
    if (rowA >= n) rowA = n - 1;              // clamp: EXEC must stay all-ones
    const float* xr = x + (long)rowA * IN_F;

    v8f acc[8] = {};

    for (int phase = 0; phase < 2; ++phase) {
        __syncthreads();
        // stage W[k = phase*128 .. +127][0..127] transposed into LDS (bf16)
        for (int i = tid; i < 128 * OUT_F; i += 256) {
            int kl = i >> 7;                  // local K
            int nc = i & 127;                 // output col
            ldsW[nc][kl] = (__bf16)w[(phase * 128 + kl) * OUT_F + nc];
        }
        __syncthreads();

        for (int kc = 0; kc < 128; kc += 32) {
            // A fragment (contiguous K order, B matches): lane holds row rowA,
            // one contiguous 16-float run at K = kc + kh*16.
            const float* p0 = xr + phase * 128 + kc + kh * 16;
            v16bf afrag;
            #pragma unroll
            for (int e2 = 0; e2 < 8; ++e2) afrag[e2]     = (__bf16)p0[e2];
            #pragma unroll
            for (int e2 = 0; e2 < 8; ++e2) afrag[8 + e2] = (__bf16)p0[8 + e2];

            // preload all 8 B fragments (16 clause-able ds_load_b128)
            v16bf bfrag[8];
            #pragma unroll
            for (int t = 0; t < 8; ++t) {
                union { v16bf v; uint4 q[2]; } bu;
                const uint4* q =
                    reinterpret_cast<const uint4*>(&ldsW[t * 16 + m16][kc + kh * 16]);
                bu.q[0] = q[0];
                bu.q[1] = q[1];
                bfrag[t] = bu.v;
            }
            // 8 back-to-back WMMAs
            #pragma unroll
            for (int t = 0; t < 8; ++t) {
                acc[t] = __builtin_amdgcn_wmma_f32_16x16x32_bf16(
                    false, afrag, false, bfrag[t], (short)0, acc[t], false, false);
            }
        }
    }

    // ---- store h. D layout: lane -> col t*16+m16; VGPR r -> row kh*8+r ----
    #pragma unroll
    for (int r = 0; r < 8; ++r) {
        int row = mBase + kh * 8 + r;
        if (row < n) {
            float* hp = h + (long)row * OUT_F + m16;
            #pragma unroll
            for (int t = 0; t < 8; ++t) hp[t * 16] = acc[t][r];
        }
    }

    // ---- fused attention scores: reduce acc over N within the wave ----
    float al[8], ar[8];
    #pragma unroll
    for (int t = 0; t < 8; ++t) {
        al[t] = a[t * 16 + m16];
        ar[t] = a[OUT_F + t * 16 + m16];
    }
    #pragma unroll
    for (int r = 0; r < 8; ++r) {
        float pl = 0.0f, pr = 0.0f;
        #pragma unroll
        for (int t = 0; t < 8; ++t) {
            float v = acc[t][r];
            pl += v * al[t];
            pr += v * ar[t];
        }
        // reduce across the 16 lanes (same kh group) holding this row's cols
        #pragma unroll
        for (int m = 1; m < 16; m <<= 1) {
            pl += __shfl_xor(pl, m, 32);
            pr += __shfl_xor(pr, m, 32);
        }
        int row = mBase + kh * 8 + r;
        if (m16 == 0 && row < n) {
            sl[row] = pl;
            sr[row] = pr;
        }
    }
}

// -------------------------------------------------------------------------
// per-edge: e = lrelu(sl[src]+sr[dst]); segment-max into emax via int-punned
// float atomic max (sign-split trick)
// -------------------------------------------------------------------------
__global__ void gat_edge_max_kernel(const int* __restrict__ src,
                                    const int* __restrict__ dst,
                                    const float* __restrict__ sl,
                                    const float* __restrict__ sr,
                                    float* __restrict__ ebuf,
                                    int* __restrict__ emax, int ne)
{
    int i = blockIdx.x * blockDim.x + threadIdx.x;
    if (i >= ne) return;
    int s = src[i], d = dst[i];
    float v = sl[s] + sr[d];
    v = (v >= 0.0f) ? v : LRELU * v;
    ebuf[i] = v;
    int iv = __float_as_int(v);
    if (iv >= 0) atomicMax(&emax[s], iv);
    else         atomicMin((unsigned int*)&emax[s], (unsigned int)iv);
}

// -------------------------------------------------------------------------
// per-edge: e_exp = exp(e - emax[src]); denom[src] += e_exp
// -------------------------------------------------------------------------
__global__ void gat_edge_exp_kernel(const int* __restrict__ src,
                                    float* __restrict__ ebuf,
                                    const int* __restrict__ emax,
                                    float* __restrict__ denom, int ne)
{
    int i = blockIdx.x * blockDim.x + threadIdx.x;
    if (i >= ne) return;
    int s = src[i];
    float m  = __int_as_float(emax[s]);
    float ex = expf(ebuf[i] - m);
    ebuf[i]  = ex;
    atomicAdd(&denom[s], ex);
}

// -------------------------------------------------------------------------
// aggregation: wave per edge; 32 lanes x float4 = 128 feats.
// out[src] += (e_exp/denom[src]) * h[dst]   (L2-resident f32 atomics;
// h = 51.2MB fits in the 192MB L2)
// -------------------------------------------------------------------------
__global__ __launch_bounds__(256) void gat_edge_agg_kernel(
    const int* __restrict__ src, const int* __restrict__ dst,
    const float* __restrict__ ebuf, const float* __restrict__ denom,
    const float* __restrict__ h, float* __restrict__ out, int ne)
{
    long gw = (long)blockIdx.x * 8 + (threadIdx.x >> 5);
    if (gw >= ne) return;
    int lane = threadIdx.x & 31;
    int s = src[gw], d = dst[gw];
    float alpha = ebuf[gw] / (denom[s] + EPS_F);
    float4 hv = ((const float4*)(h + (long)d * OUT_F))[lane];
    float* op = out + (long)s * OUT_F + lane * 4;
    atomicAdd(op + 0, alpha * hv.x);
    atomicAdd(op + 1, alpha * hv.y);
    atomicAdd(op + 2, alpha * hv.z);
    atomicAdd(op + 3, alpha * hv.w);
}

// -------------------------------------------------------------------------
extern "C" void kernel_launch(void* const* d_in, const int* in_sizes, int n_in,
                              void* d_out, int out_size, void* d_ws, size_t ws_size,
                              hipStream_t stream)
{
    const float* x  = (const float*)d_in[0];
    const int*   ei = (const int*)  d_in[1];
    const float* w  = (const float*)d_in[2];
    const float* a  = (const float*)d_in[3];
    float* out = (float*)d_out;

    const int n  = in_sizes[0] / IN_F;   // 100000 nodes
    const int ne = in_sizes[1] / 2;      // 3200000 edges
    const int* src = ei;
    const int* dst = ei + ne;

    // workspace layout (all 16B aligned)
    char* ws = (char*)d_ws;
    float* h     = (float*)ws;  ws += (size_t)n  * OUT_F * sizeof(float);
    float* sl    = (float*)ws;  ws += (size_t)n  * sizeof(float);
    float* sr    = (float*)ws;  ws += (size_t)n  * sizeof(float);
    float* ebuf  = (float*)ws;  ws += (size_t)ne * sizeof(float);
    int*   emax  = (int*)  ws;  ws += (size_t)n  * sizeof(float);
    float* denom = (float*)ws;

    const int out_elems = n * OUT_F;

    gat_init_kernel<<<(out_elems + 255) / 256, 256, 0, stream>>>(
        out, out_elems, emax, denom, n);

    gat_gemm_kernel<<<(n + 127) / 128, 256, 0, stream>>>(
        x, w, a, h, sl, sr, n);

    gat_edge_max_kernel<<<(ne + 255) / 256, 256, 0, stream>>>(
        src, dst, sl, sr, ebuf, emax, ne);

    gat_edge_exp_kernel<<<(ne + 255) / 256, 256, 0, stream>>>(
        src, ebuf, emax, denom, ne);

    gat_edge_agg_kernel<<<(ne + 7) / 8, 256, 0, stream>>>(
        src, dst, ebuf, denom, h, out, ne);
}
